// ifft_15040975471242
// MI455X (gfx1250) — matrix operand order repaired and checked by reading, not compiled, not run on hardware
//
#include <hip/hip_runtime.h>
#include <hip/hip_bf16.h>

typedef __bf16 bf16_t;
typedef __attribute__((ext_vector_type(16))) __bf16 v16bf;
typedef __attribute__((ext_vector_type(8)))  __bf16 v8bf;
typedef __attribute__((ext_vector_type(8)))  float  v8f;
typedef __attribute__((ext_vector_type(4)))  float  v4f;

#define NFFT   1024
#define NFREQ  512
#define NT     4096
#define NBATCH 8

#define BM 128
#define BN 128
#define BK 32
#define BSTR 40   // padded LDS row stride (bf16 elems) for bank spread

__device__ __forceinline__ void split2(float x, bf16_t& h, bf16_t& l) {
    h = (bf16_t)x;                 // RNE f32->bf16
    l = (bf16_t)(x - (float)h);    // residual captures next ~8 bits
}

__device__ __forceinline__ unsigned pack2(bf16_t a, bf16_t b) {
    union { bf16_t h[2]; unsigned u; } p;
    p.h[0] = a; p.h[1] = b;
    return p.u;
}

// ---------------------------------------------------------------------------
// Prep: A_comb[c][k] = (k<512 ? RK[c][k] : -IK[c][k-512]) split into hi/lo bf16
// ---------------------------------------------------------------------------
__global__ __launch_bounds__(256)
void prep_a_kernel(const float* __restrict__ rk, const float* __restrict__ ik,
                   bf16_t* __restrict__ Ahi, bf16_t* __restrict__ Alo) {
    int idx = blockIdx.x * 256 + threadIdx.x;      // 0 .. 1024*1024-1
    int c = idx >> 10;
    int k = idx & 1023;
    float v = (k < NFREQ) ? rk[c * NFREQ + k] : -ik[c * NFREQ + (k - NFREQ)];
    bf16_t h, l;
    split2(v, h, l);
    Ahi[idx] = h;
    Alo[idx] = l;
}

// ---------------------------------------------------------------------------
// Main GEMM: out[b][c][t] = sum_k A_comb[c][k] * X[b][k][t] / NFFT
//   X rows 0..511   -> magn[b],  rows 512..1023 -> phase[b]
// Block: 256 threads (8 waves), tile BM x BN, K-step BK=32.
// Wave (wm,wn in 4x2) computes 32x64 = 2x4 WMMA tiles, 3 split terms each.
// Double-buffered LDS B tile -> single barrier per K-step; next tile's global
// loads are issued before the barrier to overlap with the WMMA block.
// ---------------------------------------------------------------------------
__global__ __launch_bounds__(256)
void ifft_wmma_kernel(const float* __restrict__ magn,
                      const float* __restrict__ phase,
                      const bf16_t* __restrict__ Ahi,
                      const bf16_t* __restrict__ Alo,
                      float* __restrict__ out) {
    __shared__ __align__(16) bf16_t sBhi[2][BN * BSTR];
    __shared__ __align__(16) bf16_t sBlo[2][BN * BSTR];

    const int tid  = threadIdx.x;
    const int wave = tid >> 5;
    const int lane = tid & 31;
    const int l16  = lane & 15;
    const int half = lane >> 4;
    const int wm   = wave >> 1;   // 0..3  (M blocks of 32)
    const int wn   = wave & 1;    // 0..1  (N blocks of 64)

    const int t0 = blockIdx.x * BN;
    const int c0 = blockIdx.y * BM;
    const int b  = blockIdx.z;

    // staging decomposition: thread -> (k-pair, 8-column group)
    const int kp = tid >> 4;      // 0..15 -> rows 2kp,2kp+1 of the K-tile
    const int n0 = (tid & 15) * 8;

    union F8 { v4f v[2]; float f[8]; };
    F8 ra, rb, na, nb;

    auto load_tile = [&](int kt, F8& a, F8& c) {
        const int g  = kt >> 4;                 // 0: magn, 1: phase
        const int kk = (kt & 15) * BK;
        const float* src = (g ? phase : magn)
                         + (size_t)b * (NFREQ * NT)
                         + (size_t)(kk + 2 * kp) * NT + t0 + n0;
        a.v[0] = *(const v4f*)(src);
        a.v[1] = *(const v4f*)(src + 4);
        c.v[0] = *(const v4f*)(src + NT);
        c.v[1] = *(const v4f*)(src + NT + 4);
    };

    v8f acc[2][4];
#pragma unroll
    for (int mi = 0; mi < 2; ++mi)
#pragma unroll
        for (int ni = 0; ni < 4; ++ni)
            acc[mi][ni] = {};

    const int rowA = c0 + wm * 32 + l16;

    load_tile(0, ra, rb);

    for (int kt = 0; kt < 32; ++kt) {
        const int buf = kt & 1;

        // ---- stage current tile to LDS (f32 -> hi/lo bf16, transposed [n][k])
#pragma unroll
        for (int j = 0; j < 8; ++j) {
            bf16_t h0, l0, h1, l1;
            split2(ra.f[j], h0, l0);
            split2(rb.f[j], h1, l1);
            const int e = (n0 + j) * BSTR + 2 * kp;
            *(unsigned*)&sBhi[buf][e] = pack2(h0, h1);
            *(unsigned*)&sBlo[buf][e] = pack2(l0, l1);
        }

        // ---- issue next tile's global loads before the barrier
        load_tile(kt + 1 < 32 ? kt + 1 : 31, na, nb);

        __syncthreads();   // single barrier per K-step (double-buffered LDS)

        // ---- A fragments: direct global bf16 loads, ISA 16-bit A layout
        const int kA = kt * BK;
        union AF { v16bf v; v8bf h[2]; };
        AF ah[2], al[2], bh[4], bl[4];
#pragma unroll
        for (int mi = 0; mi < 2; ++mi) {
            const size_t ro = (size_t)(rowA + mi * 16) * NFFT + kA + half * 8;
            const bf16_t* ap = Ahi + ro;
            const bf16_t* lp = Alo + ro;
            ah[mi].h[0] = *(const v8bf*)(ap);
            ah[mi].h[1] = *(const v8bf*)(ap + 16);
            al[mi].h[0] = *(const v8bf*)(lp);
            al[mi].h[1] = *(const v8bf*)(lp + 16);
        }
        // ---- B fragments from LDS (N = l16, K = 8*half + {0..7, 16..23}, same as A)
#pragma unroll
        for (int ni = 0; ni < 4; ++ni) {
            const int e = (wn * 64 + ni * 16 + l16) * BSTR + half * 8;
            bh[ni].h[0] = *(const v8bf*)&sBhi[buf][e];
            bh[ni].h[1] = *(const v8bf*)&sBhi[buf][e + 16];
            bl[ni].h[0] = *(const v8bf*)&sBlo[buf][e];
            bl[ni].h[1] = *(const v8bf*)&sBlo[buf][e + 16];
        }

        // ---- 2x4 tiles x 3 split terms = 24 WMMAs per K-step
#pragma unroll
        for (int mi = 0; mi < 2; ++mi)
#pragma unroll
            for (int ni = 0; ni < 4; ++ni) {
                v8f c = acc[mi][ni];
                c = __builtin_amdgcn_wmma_f32_16x16x32_bf16(
                        false, ah[mi].v, false, bh[ni].v, (short)0, c, false, false);
                c = __builtin_amdgcn_wmma_f32_16x16x32_bf16(
                        false, ah[mi].v, false, bl[ni].v, (short)0, c, false, false);
                c = __builtin_amdgcn_wmma_f32_16x16x32_bf16(
                        false, al[mi].v, false, bh[ni].v, (short)0, c, false, false);
                acc[mi][ni] = c;
            }

        ra = na; rb = nb;
    }

    // ---- epilogue: scale by 1/NFFT, store per C/D layout (VGPR r -> row r)
    const float scale = 1.0f / (float)NFFT;
#pragma unroll
    for (int mi = 0; mi < 2; ++mi)
#pragma unroll
        for (int ni = 0; ni < 4; ++ni) {
            const int c = c0 + wm * 32 + mi * 16 + half * 8;
            const int t = t0 + wn * 64 + ni * 16 + l16;
            float* op = out + ((size_t)b * NFFT + c) * NT + t;
            union { v8f v; float f[8]; } u;
            u.v = acc[mi][ni];
#pragma unroll
            for (int r = 0; r < 8; ++r)
                op[(size_t)r * NT] = u.f[r] * scale;
        }
}

// ---------------------------------------------------------------------------
extern "C" void kernel_launch(void* const* d_in, const int* in_sizes, int n_in,
                              void* d_out, int out_size, void* d_ws, size_t ws_size,
                              hipStream_t stream) {
    const float* magn  = (const float*)d_in[0];   // (8,1,512,4096)
    const float* phase = (const float*)d_in[1];   // (8,1,512,4096)
    const float* rk    = (const float*)d_in[2];   // (1024,512)
    const float* ik    = (const float*)d_in[3];   // (1024,512)
    float* out = (float*)d_out;                   // (8,1024,1,4096)

    bf16_t* Ahi = (bf16_t*)d_ws;                  // 1024*1024 bf16 = 2 MB
    bf16_t* Alo = Ahi + (size_t)NFFT * NFFT;      // +2 MB  (ws usage: 4 MB)

    prep_a_kernel<<<(NFFT * NFFT) / 256, 256, 0, stream>>>(rk, ik, Ahi, Alo);

    dim3 grid(NT / BN, NFFT / BM, NBATCH);        // 32 x 8 x 8
    ifft_wmma_kernel<<<grid, 256, 0, stream>>>(magn, phase, Ahi, Alo, out);
}
